// DetectorKmeans_652835029281
// MI455X (gfx1250) — compile-verified, matrix operand order checked by reading, not run on hardware
//
#include <hip/hip_runtime.h>

// ---------------------------------------------------------------------------
// DetectorKmeans on MI455X (gfx1250):
//   cross = X @ C^T via bf16x3 split WMMA (v_wmma_f32_16x16x32_bf16)
//   sqdist = x_sq - 2*cross + c_sq   (x_sq / c_sq in exact fp32)
//   out[n] = sum_k (pr*var)[k] / sqdist[n,k] - threshold
//
// Per block: 4 wave32, 64 X-rows. Per wave: 16 rows x 32 centers per chunk
// (two 16x16 WMMA tiles -> two independent accumulator chains).
// Center chunks double-pumped with GLOBAL_LOAD_ASYNC_TO_LDS_B128 and
// progressive S_WAIT_ASYNCCNT. The chunk GEMM is fully unrolled straight-line
// code addressing LDS through two laundered addrspace(3) bases + immediates.
// ---------------------------------------------------------------------------

typedef __bf16 bf16_t;
typedef __attribute__((ext_vector_type(16))) __bf16 v16bf;
typedef __attribute__((ext_vector_type(8)))  float  v8f;
typedef __attribute__((ext_vector_type(4)))  float  f32x4;
typedef __attribute__((address_space(3))) const f32x4* lds_f4p;

#define DK_D    512      // feature dim
#define ROWS    64       // X rows per block (4 waves x 16 rows)
#define THREADS 128      // 4 wave32
#define KCH     32       // centers per K-chunk (two WMMA tiles wide)

// Row-interleaved bf16 layout: each row = [512 hi | 512 lo] = 1024 bf16 = 2KB.
#define ROWELTS   1024
#define LO        512    // element offset of the lo plane within a row

// LDS byte layout (dynamic shared base == LDS offset 0: no static __shared__)
#define XBLK_OFF  0u
#define CBLK_OFF  (XBLK_OFF + (unsigned)(ROWS * ROWELTS * 2))   // 131072
#define CRAW_OFF  (CBLK_OFF + (unsigned)(KCH * ROWELTS * 2))    // 196608, [KCH][D] f32
#define XSQ_OFF   (CRAW_OFF + (unsigned)(KCH * DK_D * 4))
#define CSQ_OFF   (XSQ_OFF + (unsigned)(ROWS * 4))
#define WGT_OFF   (CSQ_OFF + (unsigned)(KCH * 4))
#define LDS_BYTES ((size_t)(WGT_OFF + KCH * 4))

union V16BF { v16bf v; f32x4 f4[2]; };
union BF4   { bf16_t b[4]; float2 f2; };

// ---- CDNA5 async global->LDS (ASYNCcnt) via inline asm ---------------------
__device__ __forceinline__ void async_b128_to_lds(unsigned lds_byte_off,
                                                  const float* gaddr) {
  asm volatile("global_load_async_to_lds_b128 %0, %1, off"
               :
               : "v"(lds_byte_off), "v"((unsigned long long)(size_t)gaddr)
               : "memory");
}
__device__ __forceinline__ void wait_ds0() {
  asm volatile("s_wait_dscnt 0x0" ::: "memory");
}

__global__ __launch_bounds__(THREADS, 1)
void detector_kmeans_wmma(const float* __restrict__ X,
                          const float* __restrict__ center,
                          const float* __restrict__ var,
                          const float* __restrict__ pr,
                          const float* __restrict__ threshold,
                          float* __restrict__ out,
                          int K)
{
  extern __shared__ char lds_raw[];
  bf16_t* Xblk = (bf16_t*)(lds_raw + XBLK_OFF);
  bf16_t* Cblk = (bf16_t*)(lds_raw + CBLK_OFF);
  float*  Craw = (float*)(lds_raw + CRAW_OFF);
  float*  xsq  = (float*)(lds_raw + XSQ_OFF);
  float*  csq  = (float*)(lds_raw + CSQ_OFF);
  float*  wgt  = (float*)(lds_raw + WGT_OFF);

  const int t       = threadIdx.x;
  const int rowBase = blockIdx.x * ROWS;

  // Each thread owns the [4t, 4t+4) float columns of every staged row, both
  // for the async writes into Craw and the reads back out — so ASYNCcnt-only
  // synchronization is safe (no cross-thread Craw traffic before barriers).
  const unsigned crawDst = CRAW_OFF + 16u * (unsigned)t;

  // ---- kick off async stream of the FIRST center chunk (overlaps X staging)
#pragma unroll
  for (int i = 0; i < KCH; ++i)
    async_b128_to_lds(crawDst + (unsigned)i * (DK_D * 4),
                      center + (size_t)i * DK_D + 4 * t);

  if (t < ROWS) xsq[t] = 0.0f;
  __syncthreads();

  // ---- stage X tile: iteration i = row i, thread t covers cols [4t, 4t+4) ----
  for (int i = 0; i < ROWS; ++i) {
    const float4 x4 = *(const float4*)(X + (size_t)(rowBase + i) * DK_D + 4 * t);
    atomicAdd(&xsq[i], x4.x * x4.x + x4.y * x4.y + x4.z * x4.z + x4.w * x4.w);
    BF4 h, l;
    h.b[0] = (bf16_t)x4.x; l.b[0] = (bf16_t)(x4.x - (float)h.b[0]);
    h.b[1] = (bf16_t)x4.y; l.b[1] = (bf16_t)(x4.y - (float)h.b[1]);
    h.b[2] = (bf16_t)x4.z; l.b[2] = (bf16_t)(x4.z - (float)h.b[2]);
    h.b[3] = (bf16_t)x4.w; l.b[3] = (bf16_t)(x4.w - (float)h.b[3]);
    *(float2*)(Xblk + (size_t)i * ROWELTS + 4 * t)      = h.f2;
    *(float2*)(Xblk + (size_t)i * ROWELTS + LO + 4 * t) = l.f2;
  }
  __syncthreads();                       // xsq final

  // ---- per-lane WMMA operand addressing (ISA 16-bit A/B layout) ----
  // lane&15 = M (A row) / N (B col) index; lane>>4 selects K sub-block (+8).
  const int wave = t >> 5;
  const int lane = t & 31;
  const int col  = lane & 15;
  const int half = lane >> 4;
  const int koff = half * 8;

  // Two independent LDS base addresses (bytes), laundered through empty asm so
  // the backend cannot CSE them into one base + per-iteration v_add rebuilds.
  unsigned aAddr = XBLK_OFF + (unsigned)(((wave * 16 + col) * ROWELTS + koff) * 2);
  unsigned bAddr = CBLK_OFF + (unsigned)((col * ROWELTS + koff) * 2);
  asm volatile("" : "+v"(aAddr));
  asm volatile("" : "+v"(bAddr));
  const lds_f4p aP = (lds_f4p)(size_t)aAddr;   // f32x4-granular (16B) indices
  const lds_f4p bP = (lds_f4p)(size_t)bAddr;

  float xs[8];
#pragma unroll
  for (int r = 0; r < 8; ++r) xs[r] = xsq[wave * 16 + koff + r];

  float dens[8];
#pragma unroll
  for (int r = 0; r < 8; ++r) dens[r] = 0.0f;

  for (int kc = 0; kc < K; kc += KCH) {
    __syncthreads();                     // all waves done with previous B chunk
    if (t < KCH) {
      csq[t] = 0.0f;
      wgt[t] = pr[kc + t] * var[kc + t];
    }
    __syncthreads();

    // ---- convert this chunk: Craw f32 -> bf16 hi/lo + csq, progressive waits
    // async loads complete in issue order: asynccnt <= 32-8q  => rows [0,8q) in
#pragma unroll
    for (int q = 0; q < 4; ++q) {
      if (q == 0) asm volatile("s_wait_asynccnt 0x18" ::: "memory");
      if (q == 1) asm volatile("s_wait_asynccnt 0x10" ::: "memory");
      if (q == 2) asm volatile("s_wait_asynccnt 0x8"  ::: "memory");
      if (q == 3) asm volatile("s_wait_asynccnt 0x0"  ::: "memory");
#pragma unroll 2
      for (int j = 0; j < KCH / 4; ++j) {
        const int i = q * (KCH / 4) + j;
        const float4 c4 = *(const float4*)(Craw + (size_t)i * DK_D + 4 * t);
        atomicAdd(&csq[i], c4.x * c4.x + c4.y * c4.y + c4.z * c4.z + c4.w * c4.w);
        BF4 h, l;
        h.b[0] = (bf16_t)c4.x; l.b[0] = (bf16_t)(c4.x - (float)h.b[0]);
        h.b[1] = (bf16_t)c4.y; l.b[1] = (bf16_t)(c4.y - (float)h.b[1]);
        h.b[2] = (bf16_t)c4.z; l.b[2] = (bf16_t)(c4.z - (float)h.b[2]);
        h.b[3] = (bf16_t)c4.w; l.b[3] = (bf16_t)(c4.w - (float)h.b[3]);
        *(float2*)(Cblk + (size_t)i * ROWELTS + 4 * t)      = h.f2;
        *(float2*)(Cblk + (size_t)i * ROWELTS + LO + 4 * t) = l.f2;
      }
    }
    wait_ds0();                          // our Craw reads retired

    // ---- kick off async stream of the NEXT chunk (overlaps the WMMAs) ----
    if (kc + KCH < K) {
      const float* src = center + (size_t)(kc + KCH) * DK_D + 4 * t;
#pragma unroll
      for (int i = 0; i < KCH; ++i)
        async_b128_to_lds(crawDst + (unsigned)i * (DK_D * 4),
                          src + (size_t)i * DK_D);
    }
    __syncthreads();                     // hi/lo planes + csq ready

    // ---- bf16x3 split GEMM over D, fully unrolled straight-line:
    // all loads are ds_load_b128 [aP|bP] + immediate (max 34816 B).
    // f32x4 index deltas: +2 = +32B (K sub-block), +64 = lo plane (+1KB),
    // +2048 = tile-1 B rows (+32KB).
    v8f acc0 = {};
    v8f acc1 = {};
#pragma unroll
    for (int s = 0; s < 16; ++s) {
      const int o = 4 * s;
      V16BF ah, al, bh0, bl0, bh1, bl1;
      ah.f4[0]  = aP[o];          ah.f4[1]  = aP[o + 2];
      al.f4[0]  = aP[o + 64];     al.f4[1]  = aP[o + 66];
      bh0.f4[0] = bP[o];          bh0.f4[1] = bP[o + 2];
      bl0.f4[0] = bP[o + 64];     bl0.f4[1] = bP[o + 66];
      bh1.f4[0] = bP[o + 2048];   bh1.f4[1] = bP[o + 2050];
      bl1.f4[0] = bP[o + 2112];   bl1.f4[1] = bP[o + 2114];
      // interleave the two tiles so consecutive WMMAs are independent
      acc0 = __builtin_amdgcn_wmma_f32_16x16x32_bf16(false, ah.v, false, bh0.v,
                                                     (short)0, acc0, false, false);
      acc1 = __builtin_amdgcn_wmma_f32_16x16x32_bf16(false, ah.v, false, bh1.v,
                                                     (short)0, acc1, false, false);
      acc0 = __builtin_amdgcn_wmma_f32_16x16x32_bf16(false, al.v, false, bh0.v,
                                                     (short)0, acc0, false, false);
      acc1 = __builtin_amdgcn_wmma_f32_16x16x32_bf16(false, al.v, false, bh1.v,
                                                     (short)0, acc1, false, false);
      acc0 = __builtin_amdgcn_wmma_f32_16x16x32_bf16(false, ah.v, false, bl0.v,
                                                     (short)0, acc0, false, false);
      acc1 = __builtin_amdgcn_wmma_f32_16x16x32_bf16(false, ah.v, false, bl1.v,
                                                     (short)0, acc1, false, false);
    }

    // ---- fold tiles into densities ----
    // D layout: lanes 0-15 hold col, M = r; lanes 16-31 hold col, M = r + 8.
    const float cs0 = csq[col];
    const float cs1 = csq[col + 16];
    const float w0  = wgt[col];
    const float w1  = wgt[col + 16];
#pragma unroll
    for (int r = 0; r < 8; ++r) {
      const float sq0 = xs[r] - 2.0f * acc0[r] + cs0;
      const float sq1 = xs[r] - 2.0f * acc1[r] + cs1;
      dens[r] += w0 / sq0 + w1 / sq1;
    }
  }

  // ---- reduce over the 16 columns held by each half-wave ----
#pragma unroll
  for (int r = 0; r < 8; ++r) {
    float v = dens[r];
    v += __shfl_xor(v, 1, 32);
    v += __shfl_xor(v, 2, 32);
    v += __shfl_xor(v, 4, 32);
    v += __shfl_xor(v, 8, 32);
    dens[r] = v;
  }

  if (col == 0) {
    const float thr = threshold[0];
    const int g = rowBase + wave * 16 + koff;   // koff = half*8 -> rows [g, g+8)
#pragma unroll
    for (int r = 0; r < 8; ++r) out[g + r] = dens[r] - thr;
  }
}

extern "C" void kernel_launch(void* const* d_in, const int* in_sizes, int n_in,
                              void* d_out, int out_size, void* d_ws, size_t ws_size,
                              hipStream_t stream) {
  (void)n_in; (void)d_ws; (void)ws_size;
  const float* X      = (const float*)d_in[0];
  const float* center = (const float*)d_in[1];
  const float* var    = (const float*)d_in[2];
  const float* pr     = (const float*)d_in[3];
  const float* thr    = (const float*)d_in[4];
  float*       out    = (float*)d_out;

  const int K = in_sizes[2];       // number of centers (1024)
  const int N = out_size;          // number of X rows (65536)

  hipFuncSetAttribute((const void*)detector_kmeans_wmma,
                      hipFuncAttributeMaxDynamicSharedMemorySize,
                      (int)LDS_BYTES);

  dim3 grid(N / ROWS);
  detector_kmeans_wmma<<<grid, THREADS, LDS_BYTES, stream>>>(
      X, center, var, pr, thr, out, K);
}
